// MonetRouter_8770323219138
// MI455X (gfx1250) — matrix operand order repaired and checked by reading, not compiled, not run on hardware
//
#include <hip/hip_runtime.h>
#include <hip/hip_bf16.h>

#define T_TOK   8192
#define HIDDEN  1024
#define HEADS   6
#define EXPERTS 512
#define NE      3072          // HEADS*EXPERTS
#define TOPK    8
#define EPS     1e-5f
#define NEGV    -1e10f

#define BM 128
#define BN 128
#define BK 32
#define LDT 40                // padded LDS row pitch in elements (80B)
#define NKT (HIDDEN / BK)     // 32 K-tiles

typedef __attribute__((ext_vector_type(16))) __bf16 v16bf;
typedef __attribute__((ext_vector_type(8)))  __bf16 v8bf;
typedef __attribute__((ext_vector_type(4)))  __bf16 v4bf;
typedef __attribute__((ext_vector_type(8)))  float  v8f;
typedef __attribute__((ext_vector_type(4)))  float  v4f;
typedef int v4i __attribute__((vector_size(16)));   // matches builtin param type

#if __has_builtin(__builtin_amdgcn_global_load_async_to_lds_b128) && \
    __has_builtin(__builtin_amdgcn_s_wait_asynccnt)
#define USE_ASYNC_LDS 1
#else
#define USE_ASYNC_LDS 0
#endif

// ------------------------------------------------------------------
// fp32 -> bf16 conversion (vectorized, grid-stride; n4 = n/4)
// ------------------------------------------------------------------
__global__ void cvt_bf16(const float* __restrict__ s, __bf16* __restrict__ d, int n4) {
  int i = blockIdx.x * blockDim.x + threadIdx.x;
  int stride = gridDim.x * blockDim.x;
  for (; i < n4; i += stride) {
    v4f v = ((const v4f*)s)[i];
    v4bf o;
    o[0] = (__bf16)v[0]; o[1] = (__bf16)v[1];
    o[2] = (__bf16)v[2]; o[3] = (__bf16)v[3];
    ((v4bf*)d)[i] = o;
  }
}

#if USE_ASYNC_LDS
__device__ __forceinline__ void async_b128(const __bf16* g, __bf16* l) {
  __builtin_amdgcn_global_load_async_to_lds_b128(
      (__attribute__((address_space(1))) v4i*)g,
      (__attribute__((address_space(3))) v4i*)l, 0, 0);
}
#endif

// ------------------------------------------------------------------
// z[gate] = x @ w[gate]^T  via v_wmma_f32_16x16x32_bf16
// block tile 128x128, BK=32, 8 waves (2x4), wave tile 64x32 (4x2 wmma tiles)
// double-buffered LDS; async global->LDS when available
// ------------------------------------------------------------------
__global__ __launch_bounds__(256)
void gemm_wmma(const __bf16* __restrict__ X,
               const __bf16* __restrict__ W1,
               const __bf16* __restrict__ W2,
               float* __restrict__ Z) {
  __shared__ __align__(16) __bf16 sA[2][BM * LDT];
  __shared__ __align__(16) __bf16 sB[2][BN * LDT];

  const int tid  = threadIdx.x;
  const int lane = tid & 31;
  const int wave = tid >> 5;     // 0..7
  const int wm   = wave >> 2;    // 0..1 -> 64-row half
  const int wn   = wave & 3;     // 0..3 -> 32-col slice
  const int hi   = lane >> 4;    // 0/1 (lane group)
  const int lrow = lane & 15;

  const int nBase = blockIdx.x * BN;
  const int mBase = blockIdx.y * BM;
  const int gate  = blockIdx.z;
  const __bf16* Wg = gate ? W2 : W1;
  float* zout = Z + (size_t)gate * T_TOK * NE;

  // per-thread tile-copy coordinates: 2 x 16B chunks per matrix per tile
  const int r0 = tid >> 2,         c0 = (tid & 3) * 8;
  const int r1 = (256 + tid) >> 2, c1 = ((256 + tid) & 3) * 8;
  const __bf16* gA0 = X  + (size_t)(mBase + r0) * HIDDEN + c0;
  const __bf16* gA1 = X  + (size_t)(mBase + r1) * HIDDEN + c1;
  const __bf16* gB0 = Wg + (size_t)(nBase + r0) * HIDDEN + c0;
  const __bf16* gB1 = Wg + (size_t)(nBase + r1) * HIDDEN + c1;
  const int la0 = r0 * LDT + c0;
  const int la1 = r1 * LDT + c1;

  const v8f vzero = {0.f,0.f,0.f,0.f,0.f,0.f,0.f,0.f};
  v8f acc[4][2];
  #pragma unroll
  for (int i = 0; i < 4; ++i)
    #pragma unroll
    for (int j = 0; j < 2; ++j) acc[i][j] = vzero;

#if USE_ASYNC_LDS
  // prologue: async-fill buffer 0 with K-tile 0 (4 x b128 per thread)
  async_b128(gA0, &sA[0][la0]);
  async_b128(gA1, &sA[0][la1]);
  async_b128(gB0, &sB[0][la0]);
  async_b128(gB1, &sB[0][la1]);
#else
  uint4 ra0 = *(const uint4*)gA0;
  uint4 ra1 = *(const uint4*)gA1;
  uint4 rb0 = *(const uint4*)gB0;
  uint4 rb1 = *(const uint4*)gB1;
  *(uint4*)&sA[0][la0] = ra0;
  *(uint4*)&sA[0][la1] = ra1;
  *(uint4*)&sB[0][la0] = rb0;
  *(uint4*)&sB[0][la1] = rb1;
#endif

  for (int it = 0; it < NKT; ++it) {
    const int p  = it & 1;
    const int kt = (it + 1) * BK;   // next tile's K offset

#if USE_ASYNC_LDS
    __builtin_amdgcn_s_wait_asynccnt(0);   // my buf[p] writes landed
    __syncthreads();                       // everyone's writes landed; buf[p^1] reads done
    if (it + 1 < NKT) {                    // overlap next fill with compute
      async_b128(gA0 + kt, &sA[p ^ 1][la0]);
      async_b128(gA1 + kt, &sA[p ^ 1][la1]);
      async_b128(gB0 + kt, &sB[p ^ 1][la0]);
      async_b128(gB1 + kt, &sB[p ^ 1][la1]);
    }
#else
    __syncthreads();                       // buf[p] ready; buf[p^1] reads done
    uint4 na0, na1, nb0, nb1;
    if (it + 1 < NKT) {                    // global loads overlap compute (loadcnt)
      na0 = *(const uint4*)(gA0 + kt);
      na1 = *(const uint4*)(gA1 + kt);
      nb0 = *(const uint4*)(gB0 + kt);
      nb1 = *(const uint4*)(gB1 + kt);
    }
#endif

    // A fragments: per-lane row M=lane&15, K = {8*hi..+7, 16+8*hi..+7}
    v16bf af[4], bfr[2];
    #pragma unroll
    for (int i = 0; i < 4; ++i) {
      const __bf16* pa = &sA[p][(wm * 64 + i * 16 + lrow) * LDT];
      v8bf lo = *(const v8bf*)(pa + hi * 8);
      v8bf hh = *(const v8bf*)(pa + 16 + hi * 8);
      af[i] = __builtin_shufflevector(lo, hh, 0,1,2,3,4,5,6,7,8,9,10,11,12,13,14,15);
    }
    // B fragments: per-lane column N=lane&15 (a w row), K = {8*hi..+7, 16+8*hi..+7}
    #pragma unroll
    for (int j = 0; j < 2; ++j) {
      const __bf16* pb = &sB[p][(wn * 32 + j * 16 + lrow) * LDT];
      v8bf lo = *(const v8bf*)(pb + hi * 8);
      v8bf hh = *(const v8bf*)(pb + 16 + hi * 8);
      bfr[j] = __builtin_shufflevector(lo, hh, 0,1,2,3,4,5,6,7,8,9,10,11,12,13,14,15);
    }

    #pragma unroll
    for (int i = 0; i < 4; ++i)
      #pragma unroll
      for (int j = 0; j < 2; ++j)
        acc[i][j] = __builtin_amdgcn_wmma_f32_16x16x32_bf16(
            false, af[i], false, bfr[j], (short)0, acc[i][j], false, false);

#if !USE_ASYNC_LDS
    if (it + 1 < NKT) {                    // fill other buffer behind the compute
      *(uint4*)&sA[p ^ 1][la0] = na0;
      *(uint4*)&sA[p ^ 1][la1] = na1;
      *(uint4*)&sB[p ^ 1][la0] = nb0;
      *(uint4*)&sB[p ^ 1][la1] = nb1;
    }
#endif
  }

  // epilogue: D layout — VGPR v holds M = v + 8*hi, N = lane&15
  #pragma unroll
  for (int i = 0; i < 4; ++i) {
    int mrow = mBase + wm * 64 + i * 16 + hi * 8;
    #pragma unroll
    for (int j = 0; j < 2; ++j) {
      int ncol = nBase + wn * 32 + j * 16 + lrow;
      float* pp = zout + (size_t)mrow * NE + ncol;
      #pragma unroll
      for (int v = 0; v < 8; ++v) pp[(size_t)v * NE] = acc[i][j][v];
    }
  }
}

// ------------------------------------------------------------------
// In-place gating: one block per (token, gate); 6 waves = one per head.
// LayerNorm over heads, exact top-8 threshold per head, masked softmax.
// ------------------------------------------------------------------
__global__ __launch_bounds__(192)
void gate_kernel(float* __restrict__ G,
                 const float* __restrict__ g1m,
                 const float* __restrict__ g2m) {
  __shared__ float sz [HEADS * EXPERTS];
  __shared__ float szn[HEADS * EXPERTS];

  const int t    = blockIdx.x;
  const int gate = blockIdx.y;
  float* row = G + (size_t)gate * T_TOK * NE + (size_t)t * NE;
  const float* gm = gate ? g2m : g1m;

  const int tid = threadIdx.x;
  const v4f* row4 = (const v4f*)row;
  for (int i = tid; i < NE / 4; i += 192) ((v4f*)sz)[i] = row4[i];
  __syncthreads();

  // head-LayerNorm (biased var over 6 heads per expert)
  for (int e = tid; e < EXPERTS; e += 192) {
    float s = 0.f;
    #pragma unroll
    for (int h = 0; h < HEADS; ++h) s += sz[h * EXPERTS + e];
    float mu = s * (1.0f / HEADS);
    float var = 0.f;
    #pragma unroll
    for (int h = 0; h < HEADS; ++h) {
      float d = sz[h * EXPERTS + e] - mu;
      var += d * d;
    }
    var *= (1.0f / HEADS);
    float rs = rsqrtf(var + EPS);
    #pragma unroll
    for (int h = 0; h < HEADS; ++h)
      szn[h * EXPERTS + e] = (sz[h * EXPERTS + e] - mu) * rs * gm[h];
  }
  __syncthreads();

  const int head = tid >> 5;   // wave id == head (6 waves)
  const int lane = tid & 31;

  float zn[16], zr[16], wk[16];
  #pragma unroll
  for (int i = 0; i < 16; ++i) {
    int e = lane + i * 32;
    zn[i] = szn[head * EXPERTS + e];
    zr[i] = sz [head * EXPERTS + e];
    wk[i] = zn[i];
  }

  // exact 8th-largest of zn: 8 rounds of wave-argmax + remove one instance
  float thresh = -3.0e38f;
  for (int r = 0; r < TOPK; ++r) {
    float m = -3.0e38f; int mi = 0;
    #pragma unroll
    for (int i = 0; i < 16; ++i)
      if (wk[i] > m) { m = wk[i]; mi = i; }
    int owner = lane;
    #pragma unroll
    for (int off = 16; off >= 1; off >>= 1) {
      float ov = __shfl_xor(m, off, 32);
      int   oo = __shfl_xor(owner, off, 32);
      if (ov > m || (ov == m && oo < owner)) { m = ov; owner = oo; }
    }
    if (lane == owner) wk[mi] = -3.0e38f;
    thresh = m;
  }

  // masked softmax of RAW logits where zn >= thresh
  float msk[16];
  float mx = -3.0e38f;
  #pragma unroll
  for (int i = 0; i < 16; ++i) {
    msk[i] = (zn[i] >= thresh) ? zr[i] : NEGV;
    mx = fmaxf(mx, msk[i]);
  }
  #pragma unroll
  for (int off = 16; off >= 1; off >>= 1) mx = fmaxf(mx, __shfl_xor(mx, off, 32));

  float ex[16];
  float s = 0.f;
  #pragma unroll
  for (int i = 0; i < 16; ++i) { ex[i] = __expf(msk[i] - mx); s += ex[i]; }
  #pragma unroll
  for (int off = 16; off >= 1; off >>= 1) s += __shfl_xor(s, off, 32);
  float inv = 1.0f / s;

  #pragma unroll
  for (int i = 0; i < 16; ++i)
    row[head * EXPERTS + lane + i * 32] = ex[i] * inv;
}

// ------------------------------------------------------------------
extern "C" void kernel_launch(void* const* d_in, const int* in_sizes, int n_in,
                              void* d_out, int out_size, void* d_ws, size_t ws_size,
                              hipStream_t stream) {
  (void)in_sizes; (void)n_in; (void)out_size; (void)ws_size;
  const float* x   = (const float*)d_in[0];
  const float* w1  = (const float*)d_in[1];
  const float* w2  = (const float*)d_in[2];
  const float* g1m = (const float*)d_in[3];
  const float* g2m = (const float*)d_in[4];
  float* out = (float*)d_out;

  __bf16* xb  = (__bf16*)d_ws;                       // 8192*1024 bf16 (16 MB)
  __bf16* w1b = xb  + (size_t)T_TOK * HIDDEN;        // 3072*1024 bf16 (6 MB)
  __bf16* w2b = w1b + (size_t)NE * HIDDEN;           // 3072*1024 bf16 (6 MB)

  const int nx4 = T_TOK * HIDDEN / 4;
  const int nw4 = NE * HIDDEN / 4;
  cvt_bf16<<<1024, 256, 0, stream>>>(x,  xb,  nx4);
  cvt_bf16<<<1024, 256, 0, stream>>>(w1, w1b, nw4);
  cvt_bf16<<<1024, 256, 0, stream>>>(w2, w2b, nw4);

  dim3 gg(NE / BN, T_TOK / BM, 2);                   // 24 x 64 x 2 blocks
  gemm_wmma<<<gg, 256, 0, stream>>>(xb, w1b, w2b, out);

  dim3 gt(T_TOK, 2);
  gate_kernel<<<gt, 192, 0, stream>>>(out, g1m, g2m);
}
